// AttentionGCN_30812095382203
// MI455X (gfx1250) — matrix operand order repaired and checked by reading, not compiled, not run on hardware
//
#include <hip/hip_runtime.h>
#include <hip/hip_bf16.h>
#include <math.h>

// ---------------------------------------------------------------------------
// AttentionGCN for MI455X (gfx1250, wave32, WMMA).
// Heavy matmuls: v_wmma_f32_16x16x32_f16, f16 operands pre-staged in memory
// (A row-major, B transposed) so fragments load as global_load_b128.
// Wave tile 32x64 (2 A-frags x 4 B-frags -> 8 WMMA per K-step) for 2x
// operand reuse vs. a 16x16 wave tile: 1.5 b128 loads per WMMA.
// Normalized adjacency An built once per timestep in f16 (2MB, L2-resident,
// reused by both GCN GEMMs) -> 126MB adjacency streamed only twice total.
// ---------------------------------------------------------------------------

typedef __attribute__((ext_vector_type(16))) _Float16 v16h;
typedef __attribute__((ext_vector_type(8)))  _Float16 v8h;   // 16-byte load
typedef __attribute__((ext_vector_type(4)))  _Float16 v4h;   // 8-byte store
typedef __attribute__((ext_vector_type(8)))  float    v8f;

#define T_   30
#define B_   4
#define N_   256
#define BN_  1024
#define H_   128
#define DIN_ 3
#define NH_  8
#define DH_  16

// ---------------------------------------------------------------------------
// mask_flat[t, b*N+n] = ego_mask[b, t, n]  (numpy bool -> 1 byte)
__global__ void mask_flat_kernel(const unsigned char* __restrict__ ego,
                                 float* __restrict__ mask_flat) {
  int idx = blockIdx.x * blockDim.x + threadIdx.x;
  if (idx >= T_ * BN_) return;
  int t = idx / BN_, j = idx % BN_;
  int b = j / N_, n = j % N_;
  mask_flat[idx] = ego[((size_t)b * T_ + t) * N_ + n] ? 1.0f : 0.0f;
}

// ---------------------------------------------------------------------------
// deg_i = m_i * sum_j A[t,i,j]*m_j + m_i ;  dinv = deg>0 ? rsqrt(deg) : 0
// One wave (32 lanes) per row; shuffle reduction (wave32).
__global__ void __launch_bounds__(256)
dinv_kernel(const float* __restrict__ adj, const float* __restrict__ mask_flat,
            float* __restrict__ dinv) {
  int wave = threadIdx.x >> 5;
  int lane = threadIdx.x & 31;
  int rowg = blockIdx.x * 8 + wave;          // 0 .. T*BN-1
  if (rowg >= T_ * BN_) return;
  int t = rowg / BN_;
  int i = rowg % BN_;
  const float* A = adj + (size_t)t * BN_ * BN_ + (size_t)i * BN_;
  const float* m = mask_flat + (size_t)t * BN_;
  float s = 0.0f;
  for (int j = lane; j < BN_; j += 32) s += A[j] * m[j];
  for (int off = 16; off > 0; off >>= 1) s += __shfl_down(s, off, 32);
  if (lane == 0) {
    float mi  = m[i];
    float deg = mi * s + mi;
    dinv[rowg] = (deg > 0.0f) ? rsqrtf(deg) : 0.0f;
  }
}

// ---------------------------------------------------------------------------
// An_h[i,j] = (A[t,i,j]*m_i*m_j + (i==j)*m_i) * dinv_i * dinv_j  (f16)
// float4 in, half4 out; one thread = 4 consecutive j.
__global__ void __launch_bounds__(256)
an_build_kernel(const float* __restrict__ adj, const float* __restrict__ mask_flat,
                const float* __restrict__ dinv_all, _Float16* __restrict__ An_h,
                int t) {
  int idx = blockIdx.x * blockDim.x + threadIdx.x;   // BN*BN/4
  if (idx >= BN_ * BN_ / 4) return;
  int i  = idx / (BN_ / 4);
  int j0 = (idx % (BN_ / 4)) * 4;
  const float* m  = mask_flat + (size_t)t * BN_;
  const float* dv = dinv_all  + (size_t)t * BN_;
  float rs = m[i] * dv[i];                            // m_i * dinv_i
  const float4 a4  = *(const float4*)(adj + (size_t)t * BN_ * BN_ + (size_t)i * BN_ + j0);
  const float4 m4  = *(const float4*)(m + j0);
  const float4 dv4 = *(const float4*)(dv + j0);
  v4h o;
  o[0] = (_Float16)((a4.x * m4.x + (j0 + 0 == i ? 1.0f : 0.0f)) * rs * dv4.x);
  o[1] = (_Float16)((a4.y * m4.y + (j0 + 1 == i ? 1.0f : 0.0f)) * rs * dv4.y);
  o[2] = (_Float16)((a4.z * m4.z + (j0 + 2 == i ? 1.0f : 0.0f)) * rs * dv4.z);
  o[3] = (_Float16)((a4.w * m4.w + (j0 + 3 == i ? 1.0f : 0.0f)) * rs * dv4.w);
  *(v4h*)(An_h + (size_t)i * BN_ + j0) = o;
}

// ---------------------------------------------------------------------------
// X0T[h, i] = (x_raw[t] @ W1)[i, h]   (f16, transposed for B-operand use)
__global__ void xw1_kernel(const float* __restrict__ x_raw,
                           const float* __restrict__ W1,
                           _Float16* __restrict__ X0T, int t) {
  int idx = blockIdx.x * blockDim.x + threadIdx.x;  // BN*H
  if (idx >= BN_ * H_) return;
  int i = idx / H_, h = idx % H_;
  const float* x = x_raw + ((size_t)t * BN_ + i) * DIN_;
  float v = x[0] * W1[0 * H_ + h] + x[1] * W1[1 * H_ + h] + x[2] * W1[2 * H_ + h];
  X0T[(size_t)h * BN_ + i] = (_Float16)v;
}

// ---------------------------------------------------------------------------
// Core WMMA GEMM: out = epilogue( Ah[M x K] @ BTh[N x K]^T )
// f16 operands, f32 accumulate. 8 waves/block arranged 4x2; wave tile 32x64
// (2 A-frags x 4 B-frags, 8 WMMA per K-step); block tile 128x128; K step 32.
// Requires M % 128 == 0, N % 128 == 0, K % 32 == 0 (true at all call sites).
// OUT_MODE: 0 = f32 row-major, 1 = f16 row-major, 2 = f16 transposed (N x M).
template <bool RELU, bool BIAS, bool ROWMASK, int OUT_MODE>
__global__ void __launch_bounds__(256)
hgemm_kernel(const _Float16* __restrict__ Ah, const _Float16* __restrict__ BTh,
             const float* __restrict__ bias, const float* __restrict__ rowmask,
             float* __restrict__ outF, _Float16* __restrict__ outH,
             int M, int N, int K) {
  int wave = threadIdx.x >> 5, lane = threadIdx.x & 31;
  int wm = wave >> 1;               // 0..3 (row group)
  int wn = wave & 1;                // 0..1 (col group)
  int rowBase = blockIdx.x * 128 + wm * 32;
  int colBase = blockIdx.y * 128 + wn * 64;
  int halfSel = lane >> 4;          // 0: lanes 0-15, 1: lanes 16-31
  int l16     = lane & 15;

  // CDNA5 16-bit A 16x32 layout: lanes 0-15 hold K 0-7 / 16-23,
  //                              lanes 16-31 hold K 8-15 / 24-31.
  const int ka0 = halfSel ? 8 : 0;
  const int ka1 = halfSel ? 24 : 16;
  // 16-bit B 32x16 layout: lanes 0-15 hold K 0-7 / 16-23, lanes 16-31 hold K 8-15 / 24-31.
  const int kb0 = halfSel * 8;

  const _Float16* Ap0 = Ah + (size_t)(rowBase + l16) * K;       // rows r..r+15
  const _Float16* Ap1 = Ap0 + (size_t)16 * K;                   // rows r+16..r+31
  const _Float16* Bp0 = BTh + (size_t)(colBase + l16) * K;      // col tile 0

  v8f acc[2][4] = {};
  for (int k0 = 0; k0 < K; k0 += 32) {
    v16h a0, a1, b[4];
    ((v8h*)&a0)[0] = *(const v8h*)(Ap0 + k0 + ka0);
    ((v8h*)&a0)[1] = *(const v8h*)(Ap0 + k0 + ka1);
    ((v8h*)&a1)[0] = *(const v8h*)(Ap1 + k0 + ka0);
    ((v8h*)&a1)[1] = *(const v8h*)(Ap1 + k0 + ka1);
#pragma unroll
    for (int c = 0; c < 4; ++c) {
      const _Float16* Bp = Bp0 + (size_t)(16 * c) * K;
      ((v8h*)&b[c])[0] = *(const v8h*)(Bp + k0 + kb0);
      ((v8h*)&b[c])[1] = *(const v8h*)(Bp + k0 + kb0 + 16);
    }
#pragma unroll
    for (int c = 0; c < 4; ++c) {
      acc[0][c] = __builtin_amdgcn_wmma_f32_16x16x32_f16(
          false, a0, false, b[c], (short)0, acc[0][c], false, false);
      acc[1][c] = __builtin_amdgcn_wmma_f32_16x16x32_f16(
          false, a1, false, b[c], (short)0, acc[1][c], false, false);
    }
  }

  // C/D 16x16 f32 layout: VGPR r -> row r + 8*halfSel, col = lane%16
#pragma unroll
  for (int c = 0; c < 4; ++c) {
    int bcol = colBase + 16 * c + l16;
    float bb = BIAS ? bias[bcol] : 0.0f;
#pragma unroll
    for (int rt = 0; rt < 2; ++rt) {
#pragma unroll
      for (int r = 0; r < 8; ++r) {
        int orow = rowBase + rt * 16 + r + halfSel * 8;
        float v = acc[rt][c][r] + bb;
        if (RELU)    v = fmaxf(v, 0.0f);
        if (ROWMASK) v *= rowmask[orow];
        if (OUT_MODE == 0)      outF[(size_t)orow * N + bcol] = v;
        else if (OUT_MODE == 1) outH[(size_t)orow * N + bcol] = (_Float16)v;
        else                    outH[(size_t)bcol * M + orow] = (_Float16)v;
      }
    }
  }
}

// ---------------------------------------------------------------------------
// Sinusoidal positional embedding pe[T x H]
__global__ void pe_kernel(float* __restrict__ pe) {
  int idx = blockIdx.x * blockDim.x + threadIdx.x;
  if (idx >= T_ * H_) return;
  int t = idx / H_, h = idx % H_;
  int pair = h >> 1;
  float d   = __expf(-(float)(2 * pair) * (logf(10000.0f) / (float)H_));
  float ang = (float)t * d;
  pe[idx] = (h & 1) ? cosf(ang) : sinf(ang);
}

// ---------------------------------------------------------------------------
// Fuse double projection, store B-transposed f16:
//   WcT[j, i] = (Wa @ Wi)[i, j] ; bc = ba @ Wi + bi
__global__ void combine_proj_kernel(const float* __restrict__ Wa, const float* __restrict__ ba,
                                    const float* __restrict__ Wi, const float* __restrict__ bi,
                                    _Float16* __restrict__ WcT, float* __restrict__ bc) {
  int idx = blockIdx.x * blockDim.x + threadIdx.x;  // H*H
  if (idx >= H_ * H_) return;
  int i = idx / H_, j = idx % H_;
  float s = 0.0f;
  for (int k = 0; k < H_; ++k) s += Wa[i * H_ + k] * Wi[k * H_ + j];
  WcT[(size_t)j * H_ + i] = (_Float16)s;
  if (i == 0) {
    float sb = bi[j];
    for (int k = 0; k < H_; ++k) sb += ba[k] * Wi[k * H_ + j];
    bc[j] = sb;
  }
}

// ---------------------------------------------------------------------------
// WT[c, r] = W[r, c] as f16 (B-operand layout for plain weight matrices)
__global__ void f32_to_f16T_kernel(const float* __restrict__ W,
                                   _Float16* __restrict__ WT, int R, int C) {
  int idx = blockIdx.x * blockDim.x + threadIdx.x;
  if (idx >= R * C) return;
  int r = idx / C, c = idx % C;
  WT[(size_t)c * R + r] = (_Float16)W[idx];
}

// ---------------------------------------------------------------------------
// xseq_h[n, t, :] = f16( hs[t, n, :] + pe[t, :] )   (A-operand layout)
__global__ void xseq_kernel(const float* __restrict__ hs, const float* __restrict__ pe,
                            _Float16* __restrict__ xseq_h) {
  size_t idx = (size_t)blockIdx.x * blockDim.x + threadIdx.x;
  if (idx >= (size_t)BN_ * T_ * H_) return;
  int h = (int)(idx % H_);
  int t = (int)((idx / H_) % T_);
  int n = (int)(idx / ((size_t)H_ * T_));
  xseq_h[idx] = (_Float16)(hs[((size_t)t * BN_ + n) * H_ + h] + pe[t * H_ + h]);
}

// ---------------------------------------------------------------------------
// Per-(node, head) attention: 30x16 q,k,v (f16 in), 30x30 softmax, f16 out.
// T=30 is too small/misshapen for 16x16x32 WMMA and is <3% of FLOPs -> VALU.
__global__ void __launch_bounds__(32)
attn_kernel(const _Float16* __restrict__ q, const _Float16* __restrict__ k,
            const _Float16* __restrict__ v, _Float16* __restrict__ o) {
  __shared__ float sq[T_][DH_], sk[T_][DH_], sv[T_][DH_], sc[T_][T_];
  int n    = blockIdx.x / NH_;
  int head = blockIdx.x % NH_;
  int lane = threadIdx.x;
  const size_t base = (size_t)n * T_ * H_ + (size_t)head * DH_;

  for (int e = lane; e < T_ * DH_; e += 32) {
    int t = e / DH_, d = e % DH_;
    size_t g = base + (size_t)t * H_ + d;
    sq[t][d] = (float)q[g];
    sk[t][d] = (float)k[g];
    sv[t][d] = (float)v[g];
  }
  __syncthreads();

  const float scale = 0.25f;  // 1/sqrt(DH=16)
  for (int e = lane; e < T_ * T_; e += 32) {
    int t = e / T_, s = e % T_;
    float acc = 0.0f;
#pragma unroll
    for (int d = 0; d < DH_; ++d) acc += sq[t][d] * sk[s][d];
    sc[t][s] = acc * scale;
  }
  __syncthreads();

  if (lane < T_) {
    float mx = -1e30f;
    for (int s = 0; s < T_; ++s) mx = fmaxf(mx, sc[lane][s]);
    float sum = 0.0f;
    for (int s = 0; s < T_; ++s) { float e = __expf(sc[lane][s] - mx); sc[lane][s] = e; sum += e; }
    float inv = 1.0f / sum;
    for (int s = 0; s < T_; ++s) sc[lane][s] *= inv;
  }
  __syncthreads();

  for (int e = lane; e < T_ * DH_; e += 32) {
    int t = e / DH_, d = e % DH_;
    float acc = 0.0f;
    for (int s = 0; s < T_; ++s) acc += sc[t][s] * sv[s][d];
    o[base + (size_t)t * H_ + d] = (_Float16)acc;
  }
}

// ---------------------------------------------------------------------------
static inline char* carve(char*& p, size_t bytes) {
  char* r = p;
  p += (bytes + 255) & ~(size_t)255;
  return r;
}

extern "C" void kernel_launch(void* const* d_in, const int* in_sizes, int n_in,
                              void* d_out, int out_size, void* d_ws, size_t ws_size,
                              hipStream_t stream) {
  (void)in_sizes; (void)n_in; (void)out_size; (void)ws_size;

  const float*         x_raw = (const float*)d_in[0];
  const float*         adj   = (const float*)d_in[1];
  const unsigned char* ego   = (const unsigned char*)d_in[2];  // numpy bool
  const float* W1 = (const float*)d_in[3],  *b1 = (const float*)d_in[4];
  const float* W2 = (const float*)d_in[5],  *b2 = (const float*)d_in[6];
  const float* Wq = (const float*)d_in[7],  *bq = (const float*)d_in[8];
  const float* Wk = (const float*)d_in[9],  *bk = (const float*)d_in[10];
  const float* Wv = (const float*)d_in[11], *bv = (const float*)d_in[12];
  const float* Wqi= (const float*)d_in[13], *bqi= (const float*)d_in[14];
  const float* Wki= (const float*)d_in[15], *bki= (const float*)d_in[16];
  const float* Wvi= (const float*)d_in[17], *bvi= (const float*)d_in[18];
  const float* Wo = (const float*)d_in[19], *bo = (const float*)d_in[20];

  // Workspace carve-up (~58 MB), every buffer 256B aligned for b128 loads.
  char* wp = (char*)d_ws;
  float*     mask_flat = (float*)carve(wp, sizeof(float) * T_ * BN_);
  float*     dinv      = (float*)carve(wp, sizeof(float) * T_ * BN_);
  float*     pe        = (float*)carve(wp, sizeof(float) * T_ * H_);
  _Float16*  WqcT = (_Float16*)carve(wp, sizeof(_Float16) * H_ * H_);
  _Float16*  WkcT = (_Float16*)carve(wp, sizeof(_Float16) * H_ * H_);
  _Float16*  WvcT = (_Float16*)carve(wp, sizeof(_Float16) * H_ * H_);
  _Float16*  W2T  = (_Float16*)carve(wp, sizeof(_Float16) * H_ * H_);
  _Float16*  WoT  = (_Float16*)carve(wp, sizeof(_Float16) * H_ * H_);
  float*     bqc  = (float*)carve(wp, sizeof(float) * H_);
  float*     bkc  = (float*)carve(wp, sizeof(float) * H_);
  float*     bvc  = (float*)carve(wp, sizeof(float) * H_);
  _Float16*  An_h = (_Float16*)carve(wp, sizeof(_Float16) * (size_t)BN_ * BN_);
  _Float16*  X0T  = (_Float16*)carve(wp, sizeof(_Float16) * BN_ * H_);
  _Float16*  H1h  = (_Float16*)carve(wp, sizeof(_Float16) * BN_ * H_);
  _Float16*  XW2T = (_Float16*)carve(wp, sizeof(_Float16) * BN_ * H_);
  float*     hs   = (float*)carve(wp, sizeof(float) * (size_t)T_ * BN_ * H_);
  _Float16*  xseq = (_Float16*)carve(wp, sizeof(_Float16) * (size_t)BN_ * T_ * H_);
  _Float16*  qh   = (_Float16*)carve(wp, sizeof(_Float16) * (size_t)BN_ * T_ * H_);
  _Float16*  kh   = (_Float16*)carve(wp, sizeof(_Float16) * (size_t)BN_ * T_ * H_);
  _Float16*  vh   = (_Float16*)carve(wp, sizeof(_Float16) * (size_t)BN_ * T_ * H_);
  _Float16*  oh   = (_Float16*)carve(wp, sizeof(_Float16) * (size_t)BN_ * T_ * H_);

  // --- setup ---------------------------------------------------------------
  mask_flat_kernel<<<(T_ * BN_ + 255) / 256, 256, 0, stream>>>(ego, mask_flat);
  dinv_kernel<<<(T_ * BN_) / 8, 256, 0, stream>>>(adj, mask_flat, dinv);
  pe_kernel<<<(T_ * H_ + 255) / 256, 256, 0, stream>>>(pe);
  combine_proj_kernel<<<(H_ * H_) / 256, 256, 0, stream>>>(Wq, bq, Wqi, bqi, WqcT, bqc);
  combine_proj_kernel<<<(H_ * H_) / 256, 256, 0, stream>>>(Wk, bk, Wki, bki, WkcT, bkc);
  combine_proj_kernel<<<(H_ * H_) / 256, 256, 0, stream>>>(Wv, bv, Wvi, bvi, WvcT, bvc);
  f32_to_f16T_kernel<<<(H_ * H_) / 256, 256, 0, stream>>>(W2, W2T, H_, H_);
  f32_to_f16T_kernel<<<(H_ * H_) / 256, 256, 0, stream>>>(Wo, WoT, H_, H_);

  // --- GCN scan over timesteps --------------------------------------------
  dim3 gGcn(BN_ / 128, H_ / 128);      // M=1024, N=128 -> (8,1)
  for (int t = 0; t < T_; ++t) {
    an_build_kernel<<<(BN_ * BN_ / 4) / 256, 256, 0, stream>>>(adj, mask_flat, dinv, An_h, t);
    xw1_kernel<<<(BN_ * H_) / 256, 256, 0, stream>>>(x_raw, W1, X0T, t);
    // H1 = relu(An @ X0 + b1)            -> f16 row-major (A of next GEMM)
    hgemm_kernel<true, true, false, 1><<<gGcn, 256, 0, stream>>>(
        An_h, X0T, b1, nullptr, nullptr, H1h, BN_, H_, BN_);
    // XW2 = H1 @ W2                      -> f16 transposed (B of next GEMM)
    hgemm_kernel<false, false, false, 2><<<gGcn, 256, 0, stream>>>(
        H1h, W2T, nullptr, nullptr, nullptr, XW2T, BN_, H_, H_);
    // hs[t] = relu(An @ XW2 + b2) * m    -> f32 row-major
    hgemm_kernel<true, true, true, 0><<<gGcn, 256, 0, stream>>>(
        An_h, XW2T, b2, mask_flat + (size_t)t * BN_,
        hs + (size_t)t * BN_ * H_, nullptr, BN_, H_, BN_);
  }

  // --- attention over time -------------------------------------------------
  xseq_kernel<<<(int)(((size_t)BN_ * T_ * H_ + 255) / 256), 256, 0, stream>>>(hs, pe, xseq);

  dim3 gProj((BN_ * T_) / 128, H_ / 128);  // M=30720, N=128 -> (240,1)
  hgemm_kernel<false, true, false, 1><<<gProj, 256, 0, stream>>>(
      xseq, WqcT, bqc, nullptr, nullptr, qh, BN_ * T_, H_, H_);
  hgemm_kernel<false, true, false, 1><<<gProj, 256, 0, stream>>>(
      xseq, WkcT, bkc, nullptr, nullptr, kh, BN_ * T_, H_, H_);
  hgemm_kernel<false, true, false, 1><<<gProj, 256, 0, stream>>>(
      xseq, WvcT, bvc, nullptr, nullptr, vh, BN_ * T_, H_, H_);

  attn_kernel<<<BN_ * NH_, 32, 0, stream>>>(qh, kh, vh, oh);

  // out = o @ Wo + bo  -> f32 row-major (BN*T, H) == flat (B,N,T,H)
  hgemm_kernel<false, true, false, 0><<<gProj, 256, 0, stream>>>(
      oh, WoT, bo, nullptr, (float*)d_out, nullptr, BN_ * T_, H_, H_);
}
